// CrossModalAttention_17806934409285
// MI455X (gfx1250) — matrix operand order repaired and checked by reading, not compiled, not run on hardware
//
#include <hip/hip_runtime.h>

#define HIDDEN 1024
#define HEADS  16
#define HDIM   64
#define SEQ    2048
#define BATCH  2
#define ROWS   (BATCH*SEQ)   // 4096

typedef __attribute__((ext_vector_type(16))) _Float16 v16h;
typedef __attribute__((ext_vector_type(8)))  float    v8f;

union Frag {
  v16h v;
  unsigned int u[8];
  _Float16 h[16];
};

__device__ __forceinline__ v8f wmma_f16(const Frag& a, const Frag& b, v8f c) {
  // D = A(16x32 f16) * B(32x16 f16) + C(16x16 f32)
  return __builtin_amdgcn_wmma_f32_16x16x32_f16(false, a.v, false, b.v, (short)0, c, false, false);
}

// ---- A fragment: 16x32 f16 tile from row-major [M x K] source, leading dim ld.
__device__ __forceinline__ void load_a_f16(Frag& f, const _Float16* base, int ld) {
  int lane = threadIdx.x & 31;
  int m  = lane & 15;
  int hh = lane >> 4;
  const unsigned int* p = reinterpret_cast<const unsigned int*>(base + (size_t)m * ld);
#pragma unroll
  for (int j = 0; j < 8; ++j) {
    int kk = (j < 4) ? (8*hh + 2*j) : (16 + 8*hh + 2*(j-4));
    f.u[j] = p[kk >> 1];
  }
}

// ---- B fragment: 32x16 f16 tile where B[k][n] = W[n][k] (W row-major [N x K], ld).
__device__ __forceinline__ void load_b_wT(Frag& f, const _Float16* base, int ld) {
  int lane = threadIdx.x & 31;
  int n  = lane & 15;
  int hh = lane >> 4;
  const unsigned int* p = reinterpret_cast<const unsigned int*>(base + (size_t)n * ld);
#pragma unroll
  for (int j = 0; j < 8; ++j) {
    int kk = (j < 4) ? (8*hh + 2*j) : (16 + 8*hh + 2*(j-4));
    f.u[j] = p[kk >> 1];
  }
}

// ---- CDNA5 async global->LDS copies (ASYNCcnt-tracked; per-lane 8/16B) ----
__device__ __forceinline__ void async_b128(const void* g, unsigned lds_off) {
  asm volatile("global_load_async_to_lds_b128 %0, %1, off" :: "v"(lds_off), "v"(g) : "memory");
}
__device__ __forceinline__ void async_b64(const void* g, unsigned lds_off) {
  asm volatile("global_load_async_to_lds_b64 %0, %1, off" :: "v"(lds_off), "v"(g) : "memory");
}
__device__ __forceinline__ unsigned lds_off(const void* p) {
  return (unsigned)(size_t)p;   // flat shared ptr low 32 bits == LDS offset
}

// ================= f32 -> f16 convert (vectorized x4) =================
__global__ void cma17806_cvt_f16(const float* __restrict__ src, _Float16* __restrict__ dst, int n4) {
  int i = blockIdx.x * blockDim.x + threadIdx.x;
  if (i >= n4) return;
  float4 vv = reinterpret_cast<const float4*>(src)[i];
  union { _Float16 h[4]; uint2 u; } o;
  o.h[0] = (_Float16)vv.x; o.h[1] = (_Float16)vv.y;
  o.h[2] = (_Float16)vv.z; o.h[3] = (_Float16)vv.w;
  reinterpret_cast<uint2*>(dst)[i] = o.u;
}

// ================= GEMM: out = X[4096x1024] * W^T[1024x1024] + bias =================
// LDS-staged, double-buffered via async global->LDS loads (ASYNCcnt).
// Block = 8 waves; block tile 32M x 256N x 32K; wave tile 16x64.
// mode 0: f16 store | mode 1: f16 transposed V^T store | mode 2: f32 + residual
#define LDP 36   // padded LDS row pitch (f16 elems; 72B spreads banks)
__global__ __launch_bounds__(256) void cma17806_gemm(
    const _Float16* __restrict__ X, const _Float16* __restrict__ W,
    const float* __restrict__ bias,
    _Float16* __restrict__ outH, float* __restrict__ outF,
    const float* __restrict__ resid, int mode) {
  const int K = HIDDEN, N = HIDDEN;
  __shared__ _Float16 sA[2][32  * LDP];   // 2 x 2.25 KB
  __shared__ _Float16 sB[2][256 * LDP];   // 2 x 18 KB

  int tid  = threadIdx.x;
  int wv   = tid >> 5;
  int lane = tid & 31;
  int mBase = blockIdx.y * 32;
  int nBase = blockIdx.x * 256;
  const _Float16* Xb = X + (size_t)mBase * K;
  const _Float16* Wb = W + (size_t)nBase * K;

  // cooperative load assignment: every wave issues exactly 5 async ops per tile
  int ar = tid >> 3, ac = tid & 7;   // A: 32 rows x 8 x 8B chunks
  int br = tid >> 2, bc = tid & 3;   // B: 64-row groups x 4 x 16B chunks

  auto issue = [&](int kt, int buf) {
    const _Float16* xa = Xb + (size_t)ar * K + kt * 32 + ac * 4;
    async_b64(xa, lds_off(&sA[buf][ar * LDP + ac * 4]));
#pragma unroll
    for (int i = 0; i < 4; ++i) {
      int row = br + 64 * i;
      const _Float16* wp = Wb + (size_t)row * K + kt * 32 + bc * 8;
      async_b128(wp, lds_off(&sB[buf][row * LDP + bc * 8]));
    }
  };

  v8f z = {0.f,0.f,0.f,0.f,0.f,0.f,0.f,0.f};
  v8f acc[4] = {z, z, z, z};
  int wm = (wv >> 2) * 16;     // wave row offset in block tile
  int wn = (wv & 3) * 64;      // wave col offset in block tile

  issue(0, 0);
  for (int kt = 0; kt < 32; ++kt) {
    int buf = kt & 1;
    if (kt + 1 < 32) {
      issue(kt + 1, buf ^ 1);
      asm volatile("s_wait_asynccnt 0x5" ::: "memory");  // tile kt complete (in-order)
    } else {
      asm volatile("s_wait_asynccnt 0x0" ::: "memory");
    }
    __syncthreads();
    Frag a; load_a_f16(a, &sA[buf][wm * LDP], LDP);
#pragma unroll
    for (int t = 0; t < 4; ++t) {
      Frag b; load_b_wT(b, &sB[buf][(wn + 16*t) * LDP], LDP);
      acc[t] = wmma_f16(a, b, acc[t]);
    }
    __syncthreads();   // everyone done reading buf before it is refilled
  }

  int m0 = mBase + wm + (lane >> 4) * 8;
  int nl = lane & 15;
#pragma unroll
  for (int t = 0; t < 4; ++t) {
    int col = nBase + wn + 16*t + nl;
    float bval = bias[col];
#pragma unroll
    for (int r = 0; r < 8; ++r) {
      int row = m0 + r;
      float vv = acc[t][r] + bval;
      if (mode == 0) {
        outH[(size_t)row * N + col] = (_Float16)vv;
      } else if (mode == 1) {
        int bb = row >> 11, s = row & (SEQ - 1);
        outH[((size_t)bb * HIDDEN + col) * SEQ + s] = (_Float16)vv;
      } else {
        outF[(size_t)row * N + col] = vv + resid[(size_t)row * N + col];
      }
    }
  }
}

// ================= fused attention: scores + softmax + P write + context =================
// grid (128, 32): x = 16-query block, y = b*16+h. Block = 8 waves; wave w owns kv slice
// [w*256, w*256+256). Two WMMA passes (operands L2-resident); the 512MB probability
// tensor is written to HBM exactly once.
__global__ __launch_bounds__(256) void cma17806_attn(
    const _Float16* __restrict__ Q, const _Float16* __restrict__ Kf,
    const _Float16* __restrict__ Vt, const float* __restrict__ temp,
    float* __restrict__ attn, _Float16* __restrict__ ctx) {
  __shared__ float s_max[8][16];
  __shared__ float s_sum[8][16];
  __shared__ _Float16 s_p[8][16 * HDIM];
  __shared__ float s_ctx[8][16 * HDIM];

  int wv = threadIdx.x >> 5, lane = threadIdx.x & 31;
  int bh = blockIdx.y, b = bh >> 4, h = bh & 15;
  int q0 = blockIdx.x * 16;
  int hh = lane >> 4;
  int nl = lane & 15;

  const _Float16* Qb = Q  + ((size_t)(b*SEQ + q0)) * HIDDEN + h * HDIM;
  const _Float16* Vb = Vt + ((size_t)(b*HIDDEN + h*HDIM)) * SEQ;
  float ts = temp[0];
  v8f z = {0.f,0.f,0.f,0.f,0.f,0.f,0.f,0.f};

  Frag aq0, aq1;
  load_a_f16(aq0, Qb, HIDDEN);
  load_a_f16(aq1, Qb + 32, HIDDEN);

  // ---------- pass 1: online per-row max & sum ----------
  float rmax[8], rsum[8];
#pragma unroll
  for (int r = 0; r < 8; ++r) { rmax[r] = -3.402823466e38f; rsum[r] = 0.f; }

  for (int c = 0; c < 4; ++c) {
    int kv0 = wv * 256 + c * 64;
    const _Float16* Kb = Kf + ((size_t)(b*SEQ + kv0)) * HIDDEN + h * HDIM;
    v8f acc[4] = {z, z, z, z};
#pragma unroll
    for (int t = 0; t < 4; ++t) {
      Frag bf;
      load_b_wT(bf, Kb + (size_t)(16*t) * HIDDEN, HIDDEN);
      acc[t] = wmma_f16(aq0, bf, acc[t]);
      load_b_wT(bf, Kb + (size_t)(16*t) * HIDDEN + 32, HIDDEN);
      acc[t] = wmma_f16(aq1, bf, acc[t]);
    }
#pragma unroll
    for (int r = 0; r < 8; ++r) {
      float v0 = acc[0][r] * ts, v1 = acc[1][r] * ts;
      float v2 = acc[2][r] * ts, v3 = acc[3][r] * ts;
      float cm = fmaxf(fmaxf(v0, v1), fmaxf(v2, v3));
#pragma unroll
      for (int mk = 1; mk < 16; mk <<= 1) cm = fmaxf(cm, __shfl_xor(cm, mk, 32));
      float nm = fmaxf(rmax[r], cm);
      float es = __expf(v0 - nm) + __expf(v1 - nm) + __expf(v2 - nm) + __expf(v3 - nm);
#pragma unroll
      for (int mk = 1; mk < 16; mk <<= 1) es += __shfl_xor(es, mk, 32);
      rsum[r] = rsum[r] * __expf(rmax[r] - nm) + es;
      rmax[r] = nm;
    }
  }

  if (nl == 0) {
#pragma unroll
    for (int r = 0; r < 8; ++r) { s_max[wv][8*hh + r] = rmax[r]; s_sum[wv][8*hh + r] = rsum[r]; }
  }
  __syncthreads();
  float gmax[8], ginv[8];
#pragma unroll
  for (int r = 0; r < 8; ++r) {
    int row = 8*hh + r;
    float M = s_max[0][row];
#pragma unroll
    for (int ww = 1; ww < 8; ++ww) M = fmaxf(M, s_max[ww][row]);
    float S = 0.f;
#pragma unroll
    for (int ww = 0; ww < 8; ++ww) S += s_sum[ww][row] * __expf(s_max[ww][row] - M);
    gmax[r] = M;
    ginv[r] = 1.0f / S;
  }

  // ---------- pass 2: recompute, emit normalized P once, accumulate P@V ----------
  v8f cacc[4] = {z, z, z, z};
  _Float16* Pst = &s_p[wv][0];
  float* Po = attn + ((size_t)bh * SEQ + q0) * SEQ;

  for (int c = 0; c < 4; ++c) {
    int kv0 = wv * 256 + c * 64;
    const _Float16* Kb = Kf + ((size_t)(b*SEQ + kv0)) * HIDDEN + h * HDIM;
    v8f acc[4] = {z, z, z, z};
#pragma unroll
    for (int t = 0; t < 4; ++t) {
      Frag bf;
      load_b_wT(bf, Kb + (size_t)(16*t) * HIDDEN, HIDDEN);
      acc[t] = wmma_f16(aq0, bf, acc[t]);
      load_b_wT(bf, Kb + (size_t)(16*t) * HIDDEN + 32, HIDDEN);
      acc[t] = wmma_f16(aq1, bf, acc[t]);
    }
#pragma unroll
    for (int t = 0; t < 4; ++t) {
#pragma unroll
      for (int r = 0; r < 8; ++r) {
        float p = __expf(acc[t][r] * ts - gmax[r]) * ginv[r];
        int row = 8*hh + r;
        Po[(size_t)row * SEQ + kv0 + 16*t + nl] = p;
        Pst[row * HDIM + 16*t + nl] = (_Float16)p;
      }
    }
#pragma unroll
    for (int t = 0; t < 4; ++t) {
      Frag a, bf;
      load_a_f16(a, Pst, HDIM);
      load_b_wT(bf, Vb + (size_t)(16*t) * SEQ + kv0, SEQ);
      cacc[t] = wmma_f16(a, bf, cacc[t]);
      load_a_f16(a, Pst + 32, HDIM);
      load_b_wT(bf, Vb + (size_t)(16*t) * SEQ + kv0 + 32, SEQ);
      cacc[t] = wmma_f16(a, bf, cacc[t]);
    }
  }

  // ---------- cross-wave context reduction ----------
#pragma unroll
  for (int t = 0; t < 4; ++t)
#pragma unroll
    for (int r = 0; r < 8; ++r)
      s_ctx[wv][(8*hh + r) * HDIM + 16*t + nl] = cacc[t][r];
  __syncthreads();

  int tid = threadIdx.x;
#pragma unroll
  for (int e = 0; e < 4; ++e) {
    int idx = tid * 4 + e;
    float s = 0.f;
#pragma unroll
    for (int ww = 0; ww < 8; ++ww) s += s_ctx[ww][idx];
    int m = idx >> 6, d = idx & 63;
    ctx[((size_t)(b*SEQ + q0 + m)) * HIDDEN + h * HDIM + d] = (_Float16)s;
  }
}

// ================= layernorm over 1024, one block per row =================
__global__ __launch_bounds__(256) void cma17806_ln(
    const float* __restrict__ y, const float* __restrict__ gamma,
    const float* __restrict__ beta, float* __restrict__ out) {
  __shared__ float red[256];
  size_t row = blockIdx.x;
  const float* x = y + row * HIDDEN;
  int tid = threadIdx.x;
  float v[4]; float s = 0.f;
#pragma unroll
  for (int i = 0; i < 4; ++i) { v[i] = x[tid + i*256]; s += v[i]; }
  red[tid] = s; __syncthreads();
  for (int st = 128; st > 0; st >>= 1) { if (tid < st) red[tid] += red[tid+st]; __syncthreads(); }
  float mu = red[0] * (1.0f / HIDDEN); __syncthreads();
  float vs = 0.f;
#pragma unroll
  for (int i = 0; i < 4; ++i) { float d = v[i] - mu; vs += d * d; }
  red[tid] = vs; __syncthreads();
  for (int st = 128; st > 0; st >>= 1) { if (tid < st) red[tid] += red[tid+st]; __syncthreads(); }
  float inv = rsqrtf(red[0] * (1.0f / HIDDEN) + 1e-5f);
#pragma unroll
  for (int i = 0; i < 4; ++i) {
    int c = tid + i*256;
    out[row * HIDDEN + c] = (v[i] - mu) * inv * gamma[c] + beta[c];
  }
}

extern "C" void kernel_launch(void* const* d_in, const int* in_sizes, int n_in,
                              void* d_out, int out_size, void* d_ws, size_t ws_size,
                              hipStream_t stream) {
  (void)in_sizes; (void)n_in; (void)out_size; (void)ws_size;
  const float* query = (const float*)d_in[0];
  const float* key   = (const float*)d_in[1];
  const float* value = (const float*)d_in[2];
  const float* Wq = (const float*)d_in[3];
  const float* bq = (const float*)d_in[4];
  const float* Wk = (const float*)d_in[5];
  const float* bk = (const float*)d_in[6];
  const float* Wv = (const float*)d_in[7];
  const float* bv = (const float*)d_in[8];
  const float* Wo = (const float*)d_in[9];
  const float* bo = (const float*)d_in[10];
  const float* gamma = (const float*)d_in[11];
  const float* beta  = (const float*)d_in[12];
  const float* temp  = (const float*)d_in[13];

  float* out_ln = (float*)d_out;
  float* attn   = out_ln + (size_t)ROWS * HIDDEN;

  char* ws = (char*)d_ws;
  _Float16* w16q = (_Float16*)(ws + ((size_t)0  << 20));
  _Float16* w16k = (_Float16*)(ws + ((size_t)2  << 20));
  _Float16* w16v = (_Float16*)(ws + ((size_t)4  << 20));
  _Float16* w16o = (_Float16*)(ws + ((size_t)6  << 20));
  _Float16* x16  = (_Float16*)(ws + ((size_t)8  << 20));
  _Float16* q16  = (_Float16*)(ws + ((size_t)16 << 20));
  _Float16* k16  = (_Float16*)(ws + ((size_t)24 << 20));
  _Float16* vt16 = (_Float16*)(ws + ((size_t)32 << 20));
  _Float16* c16  = (_Float16*)(ws + ((size_t)40 << 20));
  float*    yf   = (float*)   (ws + ((size_t)48 << 20));

  const int W4 = (HIDDEN * HIDDEN) / 4;
  const int A4 = (ROWS * HIDDEN) / 4;
  dim3 ggrid(4, 128);
  dim3 agrid(128, 32);

  cma17806_cvt_f16<<<W4/256, 256, 0, stream>>>(Wq, w16q, W4);
  cma17806_cvt_f16<<<W4/256, 256, 0, stream>>>(Wk, w16k, W4);
  cma17806_cvt_f16<<<W4/256, 256, 0, stream>>>(Wv, w16v, W4);
  cma17806_cvt_f16<<<W4/256, 256, 0, stream>>>(Wo, w16o, W4);

  cma17806_cvt_f16<<<A4/256, 256, 0, stream>>>(query, x16, A4);
  cma17806_gemm<<<ggrid, 256, 0, stream>>>(x16, w16q, bq, q16, nullptr, nullptr, 0);
  cma17806_cvt_f16<<<A4/256, 256, 0, stream>>>(key, x16, A4);
  cma17806_gemm<<<ggrid, 256, 0, stream>>>(x16, w16k, bk, k16, nullptr, nullptr, 0);
  cma17806_cvt_f16<<<A4/256, 256, 0, stream>>>(value, x16, A4);
  cma17806_gemm<<<ggrid, 256, 0, stream>>>(x16, w16v, bv, vt16, nullptr, nullptr, 1);

  cma17806_attn<<<agrid, 256, 0, stream>>>(q16, k16, vt16, temp, attn, c16);

  cma17806_gemm<<<ggrid, 256, 0, stream>>>(c16, w16o, bo, nullptr, yf, query, 2);
  cma17806_ln<<<ROWS, 256, 0, stream>>>(yf, gamma, beta, out_ln);
}
